// GraphClassAttention_51470888075489
// MI455X (gfx1250) — matrix operand order repaired and checked by reading, not compiled, not run on hardware
//
#include <hip/hip_runtime.h>
#include <hip/hip_bf16.h>
#include <cstdint>
#include <cstddef>

typedef __bf16 v16bf  __attribute__((ext_vector_type(16)));
typedef __bf16 bf16x8 __attribute__((ext_vector_type(8)));
typedef __bf16 bf16x4 __attribute__((ext_vector_type(4)));
typedef float  v8f    __attribute__((ext_vector_type(8)));

#define BM 128
#define BN 128
#define BK 32
#define LDST 40   // BK + 8 pad elements -> 80B row stride (multiple of 16B)

union Frag { v16bf v; bf16x8 h[2]; };

struct BF2 { __bf16 hi, lo; };
__device__ __forceinline__ BF2 cvt_split(float f) {
    BF2 r;
    r.hi = (__bf16)f;
    r.lo = (__bf16)(f - (float)r.hi);
    return r;
}

// C[M x N] = alpha * (A[M x K] @ W[N x K]^T) + bias
// A rows < split come from A0, rows >= split from A1 (handles z = concat(cls, x)).
// M % 128 == 0, N % 128 == 0, K % 32 == 0 (guaranteed by caller).
__global__ __launch_bounds__(256) void gemm_bf16x3(
    const float* __restrict__ A0, const float* __restrict__ A1, int split,
    const float* __restrict__ W, const float* __restrict__ bias,
    float* __restrict__ C, int M, int N, int K, float alpha)
{
    __shared__ __bf16 Ah[BM][LDST];
    __shared__ __bf16 Al[BM][LDST];
    __shared__ __bf16 Bh[BN][LDST];
    __shared__ __bf16 Bl[BN][LDST];

    const int tid  = threadIdx.x;
    const int lane = tid & 31;
    const int wid  = tid >> 5;    // 0..7
    const int wm   = wid >> 1;    // 0..3  (M direction)
    const int wn   = wid & 1;     // 0..1  (N direction)
    const int half = lane >> 4;   // 0/1
    const int lr   = lane & 15;
    const int m0   = blockIdx.y * BM;
    const int n0   = blockIdx.x * BN;

    v8f acc[2][4];
    #pragma unroll
    for (int i = 0; i < 2; ++i)
        #pragma unroll
        for (int j = 0; j < 4; ++j)
            #pragma unroll
            for (int r = 0; r < 8; ++r) acc[i][j][r] = 0.0f;

    for (int k0 = 0; k0 < K; k0 += BK) {
        // ---- stage fp32 tiles -> LDS as hi/lo bf16 (coalesced float4 loads) ----
        #pragma unroll
        for (int i = 0; i < 4; ++i) {
            int id  = tid + i * 256;       // 0..1023 float4 slots
            int row = id >> 3;             // 0..127
            int c4  = (id & 7) << 2;       // float column offset 0..28

            int gr = m0 + row;
            const float* ap = (gr < split) ? (A0 + (size_t)gr * K)
                                           : (A1 + (size_t)(gr - split) * K);
            float4 f = *(const float4*)(ap + k0 + c4);
            bf16x4 h, l;
            {
                BF2 s0 = cvt_split(f.x); h[0] = s0.hi; l[0] = s0.lo;
                BF2 s1 = cvt_split(f.y); h[1] = s1.hi; l[1] = s1.lo;
                BF2 s2 = cvt_split(f.z); h[2] = s2.hi; l[2] = s2.lo;
                BF2 s3 = cvt_split(f.w); h[3] = s3.hi; l[3] = s3.lo;
            }
            *(bf16x4*)&Ah[row][c4] = h;
            *(bf16x4*)&Al[row][c4] = l;

            const float* bp = W + (size_t)(n0 + row) * K;
            float4 g = *(const float4*)(bp + k0 + c4);
            {
                BF2 s0 = cvt_split(g.x); h[0] = s0.hi; l[0] = s0.lo;
                BF2 s1 = cvt_split(g.y); h[1] = s1.hi; l[1] = s1.lo;
                BF2 s2 = cvt_split(g.z); h[2] = s2.hi; l[2] = s2.lo;
                BF2 s3 = cvt_split(g.w); h[3] = s3.hi; l[3] = s3.lo;
            }
            *(bf16x4*)&Bh[row][c4] = h;
            *(bf16x4*)&Bl[row][c4] = l;
        }
        __syncthreads();

        // ---- A fragments: 16x32 bf16, ISA layout:
        // lanes 0-15 (half=0): elems 0..7 = K 0..7,  elems 8..15 = K 16..23
        // lanes16-31 (half=1): elems 0..7 = K 8..15, elems 8..15 = K 24..31
        Frag a_hi[2], a_lo[2];
        #pragma unroll
        for (int mt = 0; mt < 2; ++mt) {
            int mb = wm * 32 + mt * 16 + lr;
            a_hi[mt].h[0] = *(const bf16x8*)&Ah[mb][half * 8];
            a_hi[mt].h[1] = *(const bf16x8*)&Ah[mb][16 + half * 8];
            a_lo[mt].h[0] = *(const bf16x8*)&Al[mb][half * 8];
            a_lo[mt].h[1] = *(const bf16x8*)&Al[mb][16 + half * 8];
        }

        // ---- B fragments: 32x16 bf16, lane = column N, elems 0..7 -> K = half*8 + i, elems 8..15 -> K = 16 + half*8 + (i-8)
        #pragma unroll
        for (int nt = 0; nt < 4; ++nt) {
            int nb = wn * 64 + nt * 16 + lr;
            Frag b_hi, b_lo;
            b_hi.h[0] = *(const bf16x8*)&Bh[nb][half * 8];
            b_hi.h[1] = *(const bf16x8*)&Bh[nb][16 + half * 8];
            b_lo.h[0] = *(const bf16x8*)&Bl[nb][half * 8];
            b_lo.h[1] = *(const bf16x8*)&Bl[nb][16 + half * 8];
            #pragma unroll
            for (int mt = 0; mt < 2; ++mt) {
                acc[mt][nt] = __builtin_amdgcn_wmma_f32_16x16x32_bf16(
                    false, a_hi[mt].v, false, b_hi.v, (short)0, acc[mt][nt], false, false);
                acc[mt][nt] = __builtin_amdgcn_wmma_f32_16x16x32_bf16(
                    false, a_hi[mt].v, false, b_lo.v, (short)0, acc[mt][nt], false, false);
                acc[mt][nt] = __builtin_amdgcn_wmma_f32_16x16x32_bf16(
                    false, a_lo[mt].v, false, b_hi.v, (short)0, acc[mt][nt], false, false);
            }
        }
        __syncthreads();
    }

    // ---- epilogue: D elem r -> row M = r + 8*half, col N = lr ----
    #pragma unroll
    for (int mt = 0; mt < 2; ++mt) {
        #pragma unroll
        for (int nt = 0; nt < 4; ++nt) {
            int col = n0 + wn * 64 + nt * 16 + lr;
            float bb = bias ? bias[col] : 0.0f;
            #pragma unroll
            for (int r = 0; r < 8; ++r) {
                int rowg = m0 + wm * 32 + mt * 16 + r + half * 8;
                C[(size_t)rowg * N + col] = acc[mt][nt][r] * alpha + bb;
            }
        }
    }
}

// One block of 128 threads per (head, graph). 129 keys: 128 nodes + 1 cls.
// k/v buffers are projections of z = concat(cls(512), x(65536)): cls rows 0..511.
__global__ __launch_bounds__(128) void attn_kernel(
    const float* __restrict__ q, const float* __restrict__ k,
    const float* __restrict__ v, float* __restrict__ o)
{
    const int h = blockIdx.x;   // 0..11
    const int b = blockIdx.y;   // 0..511
    const int t = threadIdx.x;  // 0..127
    const size_t LD = 768;
    const int base = h * 64;

    __shared__ float qs[64];
    __shared__ float sc[130];
    __shared__ float red[128];

    if (t < 64) qs[t] = q[(size_t)b * LD + base + t];
    __syncthreads();

    // scores: thread t -> node t; thread 0 also does cls (index 128)
    {
        size_t row = 512u + (size_t)b * 128 + t;
        const float* kp = k + row * LD + base;
        float s = 0.0f;
        #pragma unroll
        for (int d = 0; d < 64; ++d) s += qs[d] * kp[d];
        sc[t] = s;
        if (t == 0) {
            const float* kc = k + (size_t)b * LD + base;
            float s2 = 0.0f;
            #pragma unroll
            for (int d = 0; d < 64; ++d) s2 += qs[d] * kc[d];
            sc[128] = s2;
        }
    }
    __syncthreads();

    // max over 129 scores
    red[t] = sc[t];
    for (int off = 64; off > 0; off >>= 1) {
        __syncthreads();
        if (t < off) red[t] = fmaxf(red[t], red[t + off]);
    }
    __syncthreads();
    float m = fmaxf(red[0], sc[128]);
    __syncthreads();   // everyone has m before sc[128]/red are overwritten

    // exp + sum
    float e = __expf(sc[t] - m);
    sc[t] = e;
    red[t] = e;
    if (t == 0) sc[128] = __expf(sc[128] - m);
    for (int off = 64; off > 0; off >>= 1) {
        __syncthreads();
        if (t < off) red[t] += red[t + off];
    }
    __syncthreads();
    float denom = red[0] + sc[128];
    __syncthreads();   // before reusing red for partials

    // weighted value sum: d = t&63, two j-interleaved halves
    const int d = t & 63, hf = t >> 6;
    float accv = 0.0f;
    for (int j = hf; j < 129; j += 2) {
        size_t row = (j < 128) ? (512u + (size_t)b * 128 + j) : (size_t)b;
        accv += sc[j] * v[row * LD + base + d];
    }
    red[hf * 64 + d] = accv;
    __syncthreads();
    if (t < 64) {
        float num = red[t] + red[64 + t];
        o[(size_t)b * LD + base + t] = num / denom;
    }
}

extern "C" void kernel_launch(void* const* d_in, const int* in_sizes, int n_in,
                              void* d_out, int out_size, void* d_ws, size_t ws_size,
                              hipStream_t stream) {
    (void)in_sizes; (void)n_in; (void)out_size; (void)ws_size;
    const float* cls = (const float*)d_in[0];   // 512 x 768
    const float* x   = (const float*)d_in[1];   // 65536 x 768
    /* d_in[2] = graph_portion (constant 128 per graph) -> layout is implicit */
    const float* Wq  = (const float*)d_in[3];
    const float* Wk  = (const float*)d_in[4];
    const float* Wv  = (const float*)d_in[5];
    const float* Wo  = (const float*)d_in[6];
    const float* bo  = (const float*)d_in[7];
    float* out = (float*)d_out;                 // 512 x 768

    const size_t zrows = 66048, dim = 768;
    float* ws   = (float*)d_ws;
    float* kbuf = ws;                            // 66048 x 768
    float* vbuf = kbuf + zrows * dim;            // 66048 x 768
    float* qbuf = vbuf + zrows * dim;            // 512 x 768
    float* abuf = qbuf + 512 * dim;              // 512 x 768

    const float scale = 0.125f;                  // 64^-0.5
    const int BIG = 1 << 30;

    // k = z @ Wk^T * scale ; v = z @ Wv^T
    gemm_bf16x3<<<dim3(6, 516), 256, 0, stream>>>(cls, x, 512, Wk, nullptr, kbuf, 66048, 768, 768, scale);
    gemm_bf16x3<<<dim3(6, 516), 256, 0, stream>>>(cls, x, 512, Wv, nullptr, vbuf, 66048, 768, 768, 1.0f);
    // q = cls @ Wq^T
    gemm_bf16x3<<<dim3(6, 4),   256, 0, stream>>>(cls, cls, BIG, Wq, nullptr, qbuf, 512, 768, 768, 1.0f);
    // segment softmax-attention
    attn_kernel<<<dim3(12, 512), 128, 0, stream>>>(qbuf, kbuf, vbuf, abuf);
    // out = attn @ Wo^T + bo
    gemm_bf16x3<<<dim3(6, 4),   256, 0, stream>>>(abuf, abuf, BIG, Wo, bo, out, 512, 768, 768, 1.0f);
}
